// LSTM_18562848653429
// MI455X (gfx1250) — matrix operand order repaired and checked by reading, not compiled, not run on hardware
//
#include <hip/hip_runtime.h>
#include <math.h>

// ---------------------------------------------------------------------------
// LSTM forward for MI455X (gfx1250, wave32, WMMA 16x16x32 bf16, f32 accum)
//   B=64, S=2048, I=256, H=512, O=256
// ---------------------------------------------------------------------------

typedef __attribute__((ext_vector_type(16))) __bf16          v16bf;
typedef __attribute__((ext_vector_type(8)))  float           v8f;
typedef __attribute__((ext_vector_type(8)))  unsigned short  v8us;

#define B_  64
#define S_  2048
#define I_  256
#define H_  512
#define O_  256

union V16 { v16bf bf; v8us h[2]; };

// Native bf16 convert: lowers to v_cvt (packed where the compiler can fuse)
__device__ __forceinline__ unsigned short f2bf(float f) {
  return __builtin_bit_cast(unsigned short, (__bf16)f);
}
__device__ __forceinline__ float bf2f(unsigned short h) {
  union { unsigned u; float f; } v; v.u = ((unsigned)h) << 16;
  return v.f;
}
__device__ __forceinline__ float sigm(float x) { return 1.0f / (1.0f + __expf(-x)); }

// A operand (16x32 bf16, ISA 7.12.2): lane half kh holds K chunks
//   elems 0..7  = K[k0 + kh*8 + j],  elems 8..15 = K[k0 + 16 + kh*8 + j]
// B operand (32x16 bf16): lane half kh holds K chunks matching A
//   elems 0..7  = K[k0 + kh*8 + j],  elems 8..15 = K[k0 + 16 + kh*8 + j]
__device__ __forceinline__ V16 load_a_lds(const unsigned short* row, int k0, int kh) {
  V16 a;
  a.h[0] = *(const v8us*)(row + k0 + kh * 8);
  a.h[1] = *(const v8us*)(row + k0 + 16 + kh * 8);
  return a;
}
__device__ __forceinline__ V16 load_b(const unsigned short* row, int k0, int kh) {
  V16 b;
  b.h[0] = *(const v8us*)(row + k0 + kh * 8);
  b.h[1] = *(const v8us*)(row + k0 + 16 + kh * 8);
  return b;
}

// ---------------------------------------------------------------------------
// Phase 0: transpose + convert f32 weights [K,N] -> bf16 [N,K]
// ---------------------------------------------------------------------------
__global__ void k_transpose_bf16(const float* __restrict__ src,
                                 unsigned short* __restrict__ dst,
                                 int K, int N) {
  int i = blockIdx.x * blockDim.x + threadIdx.x;
  if (i >= K * N) return;
  int n = i / K;
  int k = i - n * K;
  dst[i] = f2bf(src[(size_t)k * N + n]);
}

// ---------------------------------------------------------------------------
// Phase 1: xg[g][b,t,:] = X[b,t,:] @ Wx_g + b_g   (4 gates, bf16 out)
// Block: (t, batch-block of 16). X tile staged once into LDS (bf16),
// 32 waves; wave w owns columns [16w,16w+16) of all four gates.
// Output stored in WMMA C tile layout:
//   off = ((((g*4+bb)*S + t)*32 + w)*32 + lane)*8
// ---------------------------------------------------------------------------
__global__ __launch_bounds__(1024)
void k_input_proj(const float* __restrict__ X,
                  const unsigned short* __restrict__ wxT,   // [4][512][256] bf16
                  const float* __restrict__ b0, const float* __restrict__ b1,
                  const float* __restrict__ b2, const float* __restrict__ b3,
                  unsigned short* __restrict__ xg) {
  __shared__ unsigned short xlds[16][264];                  // 16 rows x 256 (+8 pad)

  const int t  = blockIdx.x & (S_ - 1);
  const int bb = blockIdx.x >> 11;          // 0..3
  const int w  = threadIdx.x >> 5;          // 0..31
  const int ln = threadIdx.x & 31;
  const int lm = ln & 15;
  const int kh = ln >> 4;
  const int n0 = w * 16;

  // Cooperative stage: 16x256 f32 -> bf16 LDS (each thread: one 16B global load)
  {
    const int r  = threadIdx.x >> 6;        // 0..15
    const int c0 = (threadIdx.x & 63) * 4;  // 0..252
    const float* xr = X + ((size_t)(bb * 16 + r) * S_ + t) * I_ + c0;
    #pragma unroll
    for (int j = 0; j < 4; ++j) xlds[r][c0 + j] = f2bf(xr[j]);
  }
  __syncthreads();

  const float* bias[4] = { b0, b1, b2, b3 };
  v8f acc[4];
  #pragma unroll
  for (int g = 0; g < 4; ++g) {
    float bv = bias[g][n0 + lm];
    #pragma unroll
    for (int v = 0; v < 8; ++v) acc[g][v] = bv;
  }

  for (int k0 = 0; k0 < I_; k0 += 32) {
    V16 a = load_a_lds(&xlds[lm][0], k0, kh);
    #pragma unroll
    for (int g = 0; g < 4; ++g) {
      const unsigned short* wrow = wxT + ((size_t)(g * H_ + n0 + lm)) * I_;
      V16 bm = load_b(wrow, k0, kh);
      acc[g] = __builtin_amdgcn_wmma_f32_16x16x32_bf16(
          false, a.bf, false, bm.bf, (short)0, acc[g], false, false);
    }
  }

  #pragma unroll
  for (int g = 0; g < 4; ++g) {
    v8us ov;
    #pragma unroll
    for (int v = 0; v < 8; ++v) ov[v] = f2bf(acc[g][v]);
    size_t off = ((((size_t)(g * 4 + bb) * S_ + t) * 32 + w) * 32 + ln) * 8;
    *(v8us*)(xg + off) = ov;
  }
}

// ---------------------------------------------------------------------------
// Phase 2: recurrence. Batch rows are independent -> 4 workgroups of 16
// batch rows each, NO inter-workgroup sync. h in LDS (bf16), c in
// accumulator-layout VGPRs. Wave w owns h columns [16w,16w+16) for all
// four gates so the elementwise update is wave-local.
// ---------------------------------------------------------------------------
__global__ __launch_bounds__(1024)
void k_recurrence(const unsigned short* __restrict__ whT,   // [4][512][512] bf16
                  const unsigned short* __restrict__ xg,    // tile layout, bf16
                  unsigned short* __restrict__ hseq,        // [B*S][512] bf16
                  float* __restrict__ Hout,                 // [B][512] f32
                  float* __restrict__ Cout) {               // [B][512] f32
  __shared__ unsigned short hlds[16][520];

  const int bb = blockIdx.x;                // 0..3
  const int w  = threadIdx.x >> 5;
  const int ln = threadIdx.x & 31;
  const int lm = ln & 15;
  const int kh = ln >> 4;
  const int n0 = w * 16;

  // t-invariant B row pointers (Wh slices stay L2-resident)
  const unsigned short* wrow[4];
  #pragma unroll
  for (int g = 0; g < 4; ++g)
    wrow[g] = whT + ((size_t)(g * H_ + n0 + lm)) * H_;

  const size_t xg_lane = (size_t)w * 32 + ln;

  for (int i = threadIdx.x; i < 16 * 520; i += 1024)
    ((unsigned short*)hlds)[i] = 0;

  float c[8];
  #pragma unroll
  for (int v = 0; v < 8; ++v) c[v] = 0.0f;

  __syncthreads();                          // zero-init visible

  for (int t = 0; t < S_; ++t) {
    // gate pre-activations: issue early so they fly under the WMMAs,
    // and prefetch next step's tiles toward the caches
    v8us xt[4];
    #pragma unroll
    for (int g = 0; g < 4; ++g) {
      size_t off = (((size_t)(g * 4 + bb) * S_ + t) * 1024 + xg_lane) * 8;
      xt[g] = *(const v8us*)(xg + off);
      if (t + 1 < S_)
        __builtin_prefetch(xg + off + 1024u * 8u, 0, 1);
    }

    v8f acc[4];
    #pragma unroll
    for (int g = 0; g < 4; ++g)
      #pragma unroll
      for (int v = 0; v < 8; ++v) acc[g][v] = 0.0f;

    for (int k0 = 0; k0 < H_; k0 += 32) {
      V16 a = load_a_lds(&hlds[lm][0], k0, kh);
      #pragma unroll
      for (int g = 0; g < 4; ++g) {
        V16 bm = load_b(wrow[g], k0, kh);
        acc[g] = __builtin_amdgcn_wmma_f32_16x16x32_bf16(
            false, a.bf, false, bm.bf, (short)0, acc[g], false, false);
      }
    }

    float hnew[8];
    #pragma unroll
    for (int v = 0; v < 8; ++v) {
      float it = sigm(acc[0][v] + bf2f(xt[0][v]));
      float ft = sigm(acc[1][v] + bf2f(xt[1][v]));
      float ot = sigm(acc[2][v] + bf2f(xt[2][v]));
      float ct = tanhf(acc[3][v] + bf2f(xt[3][v]));
      c[v]    = ft * c[v] + it * ct;
      hnew[v] = ot * tanhf(c[v]);
    }

    __syncthreads();                        // all reads of h(t-1) done

    #pragma unroll
    for (int v = 0; v < 8; ++v) {
      const int m = v + 8 * kh;
      const int n = n0 + lm;
      hlds[m][n] = f2bf(hnew[v]);
      if (t == S_ - 1) {
        Hout[(size_t)(bb * 16 + m) * H_ + n] = hnew[v];
        Cout[(size_t)(bb * 16 + m) * H_ + n] = c[v];
      }
    }

    __syncthreads();                        // h(t) complete in LDS

    // cooperative coalesced copy LDS -> hseq (one b128 ds load + store/thread)
    {
      const int r  = threadIdx.x >> 6;      // 0..15
      const int c8 = (threadIdx.x & 63) * 8;
      v8us hv = *(const v8us*)&hlds[r][c8];
      *(v8us*)&hseq[((size_t)(bb * 16 + r) * S_ + t) * H_ + c8] = hv;
    }
  }
}

// ---------------------------------------------------------------------------
// Phase 3: out[b,t,:] = hseq[b,t,:] @ Whq + bq.  M = B*S, K = 512, N = 256.
// A tile staged through LDS once; 16 waves, wave w owns columns [16w,16w+16).
// ---------------------------------------------------------------------------
__global__ __launch_bounds__(512)
void k_output_proj(const unsigned short* __restrict__ hseq,  // [B*S][512] bf16
                   const unsigned short* __restrict__ whqT,  // [256][512] bf16
                   const float* __restrict__ bq,
                   float* __restrict__ out) {                // [B*S][256] f32
  __shared__ unsigned short alds[16][520];

  const int blk = blockIdx.x;               // 0..8191 (16 rows each)
  const int w   = threadIdx.x >> 5;         // 0..15
  const int ln  = threadIdx.x & 31;
  const int lm  = ln & 15;
  const int kh  = ln >> 4;
  const int o0  = w * 16;

  // stage A tile: 16 rows x 512 bf16 (each thread: one 32B chunk)
  {
    const int r  = threadIdx.x >> 5;        // 0..15
    const int c0 = (threadIdx.x & 31) * 16; // 0..496
    const unsigned short* src = hseq + (size_t)(blk * 16 + r) * H_ + c0;
    *(v8us*)&alds[r][c0]     = *(const v8us*)(src);
    *(v8us*)&alds[r][c0 + 8] = *(const v8us*)(src + 8);
  }
  __syncthreads();

  v8f acc;
  {
    float bv = bq[o0 + lm];
    #pragma unroll
    for (int v = 0; v < 8; ++v) acc[v] = bv;
  }

  const unsigned short* wrow = whqT + (size_t)(o0 + lm) * H_;
  for (int k0 = 0; k0 < H_; k0 += 32) {
    V16 a  = load_a_lds(&alds[lm][0], k0, kh);
    V16 bm = load_b(wrow, k0, kh);
    acc = __builtin_amdgcn_wmma_f32_16x16x32_bf16(
        false, a.bf, false, bm.bf, (short)0, acc, false, false);
  }

  #pragma unroll
  for (int v = 0; v < 8; ++v)
    out[(size_t)(blk * 16 + v + 8 * kh) * O_ + o0 + lm] = acc[v];
}

// ---------------------------------------------------------------------------
// Host launcher
// ---------------------------------------------------------------------------
extern "C" void kernel_launch(void* const* d_in, const int* in_sizes, int n_in,
                              void* d_out, int out_size, void* d_ws, size_t ws_size,
                              hipStream_t stream) {
  (void)in_sizes; (void)n_in; (void)out_size; (void)ws_size;

  const float* X      = (const float*)d_in[0];
  const float* Wx[4]  = { (const float*)d_in[1], (const float*)d_in[4],
                          (const float*)d_in[7], (const float*)d_in[10] };
  const float* Wh[4]  = { (const float*)d_in[2], (const float*)d_in[5],
                          (const float*)d_in[8], (const float*)d_in[11] };
  const float* bs[4]  = { (const float*)d_in[3], (const float*)d_in[6],
                          (const float*)d_in[9], (const float*)d_in[12] };
  const float* Whq    = (const float*)d_in[13];
  const float* bq     = (const float*)d_in[14];

  // Workspace layout (bytes):
  //   [0, 1MB)        wxT  : 4*512*256 bf16
  //   [1MB, 3MB)      whT  : 4*512*512 bf16
  //   [3MB, 3.25MB)   whqT : 256*512   bf16
  //   [4MB, 516MB)    xg   : 4*64*2048*512 bf16 (WMMA tile layout)
  //   [516MB, 644MB)  hseq : 64*2048*512 bf16
  char* ws = (char*)d_ws;
  unsigned short* wxT  = (unsigned short*)(ws);
  unsigned short* whT  = (unsigned short*)(ws + (1ull << 20));
  unsigned short* whqT = (unsigned short*)(ws + (3ull << 20));
  unsigned short* xg   = (unsigned short*)(ws + (4ull << 20));
  unsigned short* hseq = (unsigned short*)(ws + (4ull << 20) + (512ull << 20));

  for (int g = 0; g < 4; ++g) {
    k_transpose_bf16<<<(I_ * H_ + 255) / 256, 256, 0, stream>>>(
        Wx[g], wxT + (size_t)g * H_ * I_, I_, H_);
    k_transpose_bf16<<<(H_ * H_ + 255) / 256, 256, 0, stream>>>(
        Wh[g], whT + (size_t)g * H_ * H_, H_, H_);
  }
  k_transpose_bf16<<<(H_ * O_ + 255) / 256, 256, 0, stream>>>(Whq, whqT, H_, O_);

  k_input_proj<<<dim3(S_ * 4), dim3(1024), 0, stream>>>(
      X, wxT, bs[0], bs[1], bs[2], bs[3], xg);

  float* out  = (float*)d_out;
  float* Hout = out + (size_t)B_ * S_ * O_;
  float* Cout = Hout + (size_t)B_ * H_;
  k_recurrence<<<dim3(4), dim3(1024), 0, stream>>>(whT, xg, hseq, Hout, Cout);

  k_output_proj<<<dim3((B_ * S_) / 16), dim3(512), 0, stream>>>(
      hseq, whqT, bq, out);
}
